// MultiInputLSTMMacroFactors_19842748908309
// MI455X (gfx1250) — matrix operand order repaired and checked by reading, not compiled, not run on hardware
//
#include <hip/hip_runtime.h>

#define BS   256
#define SEQ  512
#define INF_ 64
#define H    256
#define KIN  10

// padded LDS row strides (in elements) to avoid bank conflicts:
// 264 ushorts = 132 words == 4 (mod 64); 72 ushorts = 36 words, 4*l distinct.
#define LDH  264
#define LDX  72

typedef __attribute__((ext_vector_type(16))) __bf16         v16bf;
typedef __attribute__((ext_vector_type(8)))  float          v8f;
typedef __attribute__((ext_vector_type(16))) unsigned short v16us;

__device__ __forceinline__ unsigned short f2bf(float f) {
  unsigned int u = __float_as_uint(f);
  u += 0x7FFFu + ((u >> 16) & 1u);
  return (unsigned short)(u >> 16);
}

__device__ __forceinline__ v8f vzero() {
  v8f z = {0.f, 0.f, 0.f, 0.f, 0.f, 0.f, 0.f, 0.f};
  return z;
}

__device__ __forceinline__ v8f wmma_bf16(v16bf a, v16bf b, v8f c) {
  return __builtin_amdgcn_wmma_f32_16x16x32_bf16(false, a, false, b, (short)0,
                                                 c, false, false);
}

// ---------------------------------------------------------------------------
// A fragment (16x32 bf16, ISA 7.12.2): lanes 0-15 row M=l, elems 0-7 =
// K k0..k0+7, elems 8-15 = K k0+16..23; lanes 16-31 K halves shifted by +8.
// ---------------------------------------------------------------------------
__device__ __forceinline__ v16bf a_frag_lds(const unsigned short* A, int ld,
                                            int k0, int lane) {
  const int l  = lane & 15;
  const int hi = (lane >> 4) & 1;
  const unsigned short* p = A + l * ld + k0 + hi * 8;
  v16us r;
#pragma unroll
  for (int j = 0; j < 8; ++j) { r[j] = p[j]; r[8 + j] = p[16 + j]; }
  return __builtin_bit_cast(v16bf, r);
}

// B fragment (32x16 bf16) from bf16 transposed weights BT[N][K]
__device__ __forceinline__ v16bf b_frag_g(const unsigned short* BT, int Kd,
                                          int n0, int k0, int lane) {
  const int l  = lane & 15;
  const int hi = (lane >> 4) & 1;
  const unsigned short* p = BT + (size_t)(n0 + l) * Kd + k0 + hi * 8;
  v16us r;
#pragma unroll
  for (int j = 0; j < 8; ++j) { r[j] = p[j]; r[8 + j] = p[16 + j]; }
  return __builtin_bit_cast(v16bf, r);
}

// K=256 GEMM with pre-loaded A fragments; dual accumulators to halve the
// WMMA->WMMA RAW chain (ISA 7.12.1 hazard).
__device__ __forceinline__ v8f gemmH(const v16bf (&af)[8],
                                     const unsigned short* BT, int n0,
                                     int lane, v8f accIn) {
  v8f a0 = accIn, a1 = vzero();
#pragma unroll
  for (int kk = 0; kk < 4; ++kk) {
    a0 = wmma_bf16(af[2 * kk],     b_frag_g(BT, 256, n0, (2 * kk) * 32, lane), a0);
    a1 = wmma_bf16(af[2 * kk + 1], b_frag_g(BT, 256, n0, (2 * kk + 1) * 32, lane), a1);
  }
  return a0 + a1;
}

// K=256 GEMM with A fragments re-read from LDS (phase 2 l-tiles)
__device__ __forceinline__ v8f gemmS(const unsigned short* A, int ldA,
                                     const unsigned short* BT, int n0,
                                     int lane, v8f accIn) {
  v8f a0 = accIn, a1 = vzero();
#pragma unroll
  for (int kk = 0; kk < 4; ++kk) {
    a0 = wmma_bf16(a_frag_lds(A, ldA, (2 * kk) * 32, lane),
                   b_frag_g(BT, 256, n0, (2 * kk) * 32, lane), a0);
    a1 = wmma_bf16(a_frag_lds(A, ldA, (2 * kk + 1) * 32, lane),
                   b_frag_g(BT, 256, n0, (2 * kk + 1) * 32, lane), a1);
  }
  return a0 + a1;
}

// K=64 GEMM with pre-loaded x fragments
__device__ __forceinline__ v8f gemmX(v16bf xa, v16bf xb,
                                     const unsigned short* BT, int n0,
                                     int lane, v8f acc) {
  acc = wmma_bf16(xa, b_frag_g(BT, 64, n0, 0, lane), acc);
  acc = wmma_bf16(xb, b_frag_g(BT, 64, n0, 32, lane), acc);
  return acc;
}

struct XPtrs { const float* p[KIN]; };

// ---------------------------------------------------------------------------
// Weight convert + transpose: src f32 [R][C] (R = reduction dim) -> bf16 [C][R]
// ---------------------------------------------------------------------------
__global__ void convT(const float* __restrict__ src,
                      unsigned short* __restrict__ dst, int R, int C) {
  const int mat = blockIdx.y;
  const float* s = src + (size_t)mat * R * C;
  unsigned short* d = dst + (size_t)mat * R * C;
  const int total = R * C;
  for (int idx = blockIdx.x * blockDim.x + threadIdx.x; idx < total;
       idx += gridDim.x * blockDim.x) {
    const int r = idx / C, c = idx % C;
    d[(size_t)c * R + r] = f2bf(s[idx]);
  }
}

// ---------------------------------------------------------------------------
// One full timestep (both phases). Grid: 16 blocks (16 batch rows each),
// 512 threads = 16 waves, one 16x16 column tile per wave.
// ---------------------------------------------------------------------------
__global__ __launch_bounds__(512) void k_step(
    XPtrs xs, int t,
    const unsigned short* __restrict__ UTi, const unsigned short* __restrict__ UTcp,
    const unsigned short* __restrict__ UTcn, const unsigned short* __restrict__ UTo,
    const unsigned short* __restrict__ UTa,
    const unsigned short* __restrict__ WTi, const unsigned short* __restrict__ WTcp,
    const unsigned short* __restrict__ WTcn, const unsigned short* __restrict__ WTo,
    const float* __restrict__ b_i, const float* __restrict__ b_cp,
    const float* __restrict__ b_cn, const float* __restrict__ b_o,
    const float* __restrict__ b_a,
    unsigned short* __restrict__ hbf, float* __restrict__ c_ws,
    float* __restrict__ l_ws, float* __restrict__ out) {
  __shared__ unsigned short sh_a[16 * LDH];        // h tile; reused for l tiles
  __shared__ unsigned short sh_x[KIN][16 * LDX];

  const int tid  = threadIdx.x;
  const int lane = tid & 31;
  const int wave = tid >> 5;          // 0..15
  const int m0   = blockIdx.x * 16;
  const int n0   = wave * 16;

  const int row  = tid >> 5;          // 0..15 (staging row)
  const int cseg = tid & 31;          // 32 threads per row

  // ---- stage h_{t-1} (bf16 from state buffer) and x_t[k] tiles ----
  {
    const unsigned short* hp = hbf + (size_t)(m0 + row) * H + cseg * 8;
#pragma unroll
    for (int j = 0; j < 8; ++j) sh_a[row * LDH + cseg * 8 + j] = hp[j];
#pragma unroll
    for (int k = 0; k < KIN; ++k) {
      const float* xp =
          xs.p[k] + ((size_t)(m0 + row) * SEQ + (size_t)t) * INF_ + cseg * 2;
      sh_x[k][row * LDX + cseg * 2 + 0] = f2bf(xp[0]);
      sh_x[k][row * LDX + cseg * 2 + 1] = f2bf(xp[1]);
    }
  }
  __syncthreads();

  // hoist h A-fragments: reused by 13 K=256 GEMMs
  v16bf hfr[8];
#pragma unroll
  for (int kk = 0; kk < 8; ++kk) hfr[kk] = a_frag_lds(sh_a, LDH, kk * 32, lane);

  const int col = n0 + (lane & 15);
  const int rl  = ((lane >> 4) & 1) * 8;    // row offset within 16-row tile

  // ---- phase 1: gates ----
  v8f hUo = gemmH(hfr, UTo, n0, lane, vzero());
  v8f st  = gemmH(hfr, UTcp, n0, lane, vzero());
  v8f rt  = gemmH(hfr, UTcn, n0, lane, vzero());
  {
    v16bf x0a = a_frag_lds(sh_x[0], LDX, 0, lane);
    v16bf x0b = a_frag_lds(sh_x[0], LDX, 32, lane);
    st = gemmX(x0a, x0b, WTcp, n0, lane, st);
    v16bf x1a = a_frag_lds(sh_x[1], LDX, 0, lane);
    v16bf x1b = a_frag_lds(sh_x[1], LDX, 32, lane);
    rt = gemmX(x1a, x1b, WTcn, n0, lane, rt);
  }
  const float bcp = b_cp[col], bcn = b_cn[col], bo = b_o[col];
#pragma unroll
  for (int r = 0; r < 8; ++r) {
    float s = st[r] + bcp; st[r] = s > 0.f ? s : 0.f;
    float q = rt[r] + bcn; rt[r] = q > 0.f ? q : 0.f;
  }

  v8f o_acc = vzero();
#pragma unroll
  for (int k = 0; k < KIN; ++k) {
    v16bf xa = a_frag_lds(sh_x[k], LDX, 0, lane);
    v16bf xb = a_frag_lds(sh_x[k], LDX, 32, lane);
    // i_g[k] = relu(x_k @ W_i[k] + h @ U_i[k] + b_i[k])
    v8f acc = gemmX(xa, xb, WTi + (size_t)k * 64 * 256, n0, lane, vzero());
    acc = gemmH(hfr, UTi + (size_t)k * 256 * 256, n0, lane, acc);
    const float bi = b_i[k * H + col];
    const v8f tl = (k == 0) ? st : rt;
#pragma unroll
    for (int r = 0; r < 8; ++r) {
      float ig = acc[r] + bi;
      ig = ig > 0.f ? ig : 0.f;
      l_ws[(size_t)k * BS * H + (size_t)(m0 + rl + r) * H + col] = tl[r] * ig;
    }
    // o_t += relu(x_k @ W_o + h @ U_o + b_o)
    v8f oc = gemmX(xa, xb, WTo, n0, lane, vzero());
#pragma unroll
    for (int r = 0; r < 8; ++r) {
      float v = oc[r] + hUo[r] + bo;
      o_acc[r] += v > 0.f ? v : 0.f;
    }
  }

  // ---- phase 2: u = relu((l@W_a)*c + b_a), softmax over k, state update ----
  const float ba = b_a[col];
  float cv[8];
#pragma unroll
  for (int r = 0; r < 8; ++r)
    cv[r] = c_ws[(size_t)(m0 + rl + r) * H + col];

  v8f u[KIN];
#pragma unroll
  for (int k = 0; k < KIN; ++k) {
    __syncthreads();   // also makes this block's l_ws stores visible
    const float* lp =
        l_ws + (size_t)k * BS * H + (size_t)(m0 + row) * H + cseg * 8;
#pragma unroll
    for (int j = 0; j < 8; ++j)
      sh_a[row * LDH + cseg * 8 + j] = f2bf(lp[j]);
    __syncthreads();
    v8f acc = gemmS(sh_a, LDH, UTa, n0, lane, vzero());
#pragma unroll
    for (int r = 0; r < 8; ++r) {
      float v = acc[r] * cv[r] + ba;
      u[k][r] = v > 0.f ? v : 0.f;
    }
  }

  float* hid = out + (size_t)BS * H;
#pragma unroll
  for (int r = 0; r < 8; ++r) {
    float m = u[0][r];
#pragma unroll
    for (int k = 1; k < KIN; ++k) m = fmaxf(m, u[k][r]);
    float e[KIN];
    float s = 0.f;
#pragma unroll
    for (int k = 0; k < KIN; ++k) { e[k] = __expf(u[k][r] - m); s += e[k]; }
    const float inv = 1.0f / s;
    const size_t eo = (size_t)(m0 + rl + r) * H + col;
    float L = 0.f;
#pragma unroll
    for (int k = 0; k < KIN; ++k)
      L += (e[k] * inv) * l_ws[(size_t)k * BS * H + eo];
    const float cn = cv[r] + L;
    c_ws[eo] = cn;
    const float hr = o_acc[r] * (cn > 0.f ? cn : 0.f);
    hid[((size_t)(m0 + rl + r) * SEQ + (size_t)t) * H + col] = hr;
    hbf[eo] = f2bf(hr);                   // bf16 state for next step
    if (t == SEQ - 1) out[eo] = hr;       // h_last
  }
}

// ---------------------------------------------------------------------------
extern "C" void kernel_launch(void* const* d_in, const int* in_sizes, int n_in,
                              void* d_out, int out_size, void* d_ws,
                              size_t ws_size, hipStream_t stream) {
  (void)in_sizes; (void)n_in; (void)out_size; (void)ws_size;
  char* ws = (char*)d_ws;
  float*          c_ws = (float*)(ws + 0);              // 262,144 B
  unsigned short* hbf  = (unsigned short*)(ws + 262144);  // 131,072 B
  float*          l_ws = (float*)(ws + 393216);         // 2,621,440 B
  unsigned short* UTi  = (unsigned short*)(ws + 3014656); // 1,310,720 B
  unsigned short* UTcp = (unsigned short*)(ws + 4325376);
  unsigned short* UTcn = (unsigned short*)(ws + 4456448);
  unsigned short* UTo  = (unsigned short*)(ws + 4587520);
  unsigned short* UTa  = (unsigned short*)(ws + 4718592);
  unsigned short* WTi  = (unsigned short*)(ws + 4849664); // 327,680 B
  unsigned short* WTcp = (unsigned short*)(ws + 5177344);
  unsigned short* WTcn = (unsigned short*)(ws + 5210112);
  unsigned short* WTo  = (unsigned short*)(ws + 5242880); // ends 5,275,648

  const float* W_i  = (const float*)d_in[10];
  const float* U_i  = (const float*)d_in[11];
  const float* b_i  = (const float*)d_in[12];
  const float* W_cp = (const float*)d_in[13];
  const float* U_cp = (const float*)d_in[14];
  const float* b_cp = (const float*)d_in[15];
  const float* W_cn = (const float*)d_in[16];
  const float* U_cn = (const float*)d_in[17];
  const float* b_cn = (const float*)d_in[18];
  const float* W_o  = (const float*)d_in[19];
  const float* U_o  = (const float*)d_in[20];
  const float* b_o  = (const float*)d_in[21];
  const float* W_a  = (const float*)d_in[22];
  const float* b_a  = (const float*)d_in[23];

  XPtrs xs;
  for (int k = 0; k < KIN; ++k) xs.p[k] = (const float*)d_in[k];

  // one-time bf16 transpose of all weights (reused 512x)
  convT<<<dim3(64, KIN), 256, 0, stream>>>(U_i, UTi, H, H);
  convT<<<dim3(64, 1),   256, 0, stream>>>(U_cp, UTcp, H, H);
  convT<<<dim3(64, 1),   256, 0, stream>>>(U_cn, UTcn, H, H);
  convT<<<dim3(64, 1),   256, 0, stream>>>(U_o, UTo, H, H);
  convT<<<dim3(64, 1),   256, 0, stream>>>(W_a, UTa, H, H);
  convT<<<dim3(16, KIN), 256, 0, stream>>>(W_i, WTi, INF_, H);
  convT<<<dim3(16, 1),   256, 0, stream>>>(W_cp, WTcp, INF_, H);
  convT<<<dim3(16, 1),   256, 0, stream>>>(W_cn, WTcn, INF_, H);
  convT<<<dim3(16, 1),   256, 0, stream>>>(W_o, WTo, INF_, H);

  hipMemsetAsync(c_ws, 0, (size_t)BS * H * sizeof(float), stream);
  hipMemsetAsync(hbf, 0, (size_t)BS * H * sizeof(unsigned short), stream);

  float* out = (float*)d_out;
  dim3 grid(BS / 16), block(512);
  for (int t = 0; t < SEQ; ++t) {
    k_step<<<grid, block, 0, stream>>>(xs, t, UTi, UTcp, UTcn, UTo, UTa, WTi,
                                       WTcp, WTcn, WTo, b_i, b_cp, b_cn, b_o,
                                       b_a, hbf, c_ws, l_ws, out);
  }
}
